// QuadAttention_47691316855186
// MI455X (gfx1250) — matrix operand order repaired and checked by reading, not compiled, not run on hardware
//
#include <hip/hip_runtime.h>
#include <hip/hip_bf16.h>

// QuadAttention for MI455X (gfx1250, wave32, WMMA).
// out  = prob @ v,  prob = (q@k^T*scale + 3)^2 / rowsum(...) , fp32 I/O.
// Strategy: per-wave 16-query tile, two sweeps over K (den pass, prob/out pass),
// both GEMMs via v_wmma_f32_16x16x32_f16; K/V staged in LDS as f16.

typedef __attribute__((ext_vector_type(16))) _Float16 v16h;
typedef __attribute__((ext_vector_type(8)))  _Float16 v8h;
typedef __attribute__((ext_vector_type(8)))  float    v8f;

#define S_DIM 2048
#define D_DIM 64
#define SCALE 0.125f
#define C_ADD 3.0f
#define EPSV  1e-6f

// LDS row strides (in halves) chosen 16B-aligned and bank-conflict-free.
#define KSTR 72   // 144 B rows for K tile [32][64]
#define VSTR 40   // 80  B rows for V^T tile [64][32] and prob scratch [16][32]

__device__ __forceinline__ v8h pack8(float4 a, float4 b) {
  v8h r;
  r[0] = (_Float16)a.x; r[1] = (_Float16)a.y; r[2] = (_Float16)a.z; r[3] = (_Float16)a.w;
  r[4] = (_Float16)b.x; r[5] = (_Float16)b.y; r[6] = (_Float16)b.z; r[7] = (_Float16)b.w;
  return r;
}

__device__ __forceinline__ v16h cat16(v8h lo, v8h hi) {
  return __builtin_shufflevector(lo, hi, 0,1,2,3,4,5,6,7,8,9,10,11,12,13,14,15);
}

__device__ __forceinline__ v8f wmma_f16(v16h a, v16h b, v8f c) {
  // D = A(16x32 f16) * B(32x16 f16) + C(16x16 f32)
  return __builtin_amdgcn_wmma_f32_16x16x32_f16(false, a, false, b, (short)0, c, false, false);
}

__global__ void __launch_bounds__(128)
QuadAttention_kernel(const float* __restrict__ q, const float* __restrict__ k,
                     const float* __restrict__ v, float* __restrict__ out,
                     float* __restrict__ prob) {
  __shared__ __align__(16) _Float16 sK[32 * KSTR];          // K tile  [j][d], f16
  __shared__ __align__(16) _Float16 sV[64 * VSTR];          // V tile^T [d][j], f16
  __shared__ __align__(16) _Float16 sP[4 * 16 * VSTR];      // per-wave prob scratch [m][j]

  const int bh   = blockIdx.y;                 // batch*head, 0..31
  const int tid  = threadIdx.x;
  const int w    = tid >> 5;                   // wave id 0..3
  const int lane = tid & 31;
  const int m    = lane & 15;                  // row/col within 16
  const int hi   = lane >> 4;                  // lane half (A/B K-run selector)
  const int q0   = blockIdx.x * 64 + w * 16;   // first query row of this wave

  const float* Q = q + (size_t)bh * S_DIM * D_DIM;
  const float* K = k + (size_t)bh * S_DIM * D_DIM;
  const float* V = v + (size_t)bh * S_DIM * D_DIM;
  float* OUT = out  + (size_t)bh * S_DIM * D_DIM;
  float* PR  = prob + (size_t)bh * S_DIM * S_DIM;

  // ---- Load Q tile into WMMA A-layout (two 16x32 K-chunks over d) ----
  // A operand in this code's k order (16x32): lane<16 holds K=0..15; lane>=16 holds K=16..31.
  v16h Qa[2];
#pragma unroll
  for (int c = 0; c < 2; ++c) {
    const float* qr = Q + (size_t)(q0 + m) * D_DIM + c * 32 + hi * 16;
    float4 a0 = *(const float4*)(qr + 0),  a1 = *(const float4*)(qr + 4);
    float4 b0 = *(const float4*)(qr + 8), b1 = *(const float4*)(qr + 12);
    Qa[c] = cat16(pack8(a0, a1), pack8(b0, b1));
  }

  const int jc = tid >> 2;        // cooperative fill: key row 0..31
  const int dg = tid & 3;         // 16-wide d group 0..3

  // ================= Sweep 1: row sums of (s*scale+3)^2 =================
  v8f den = {};
  for (int j0 = 0; j0 < S_DIM; j0 += 32) {
    {  // stage K tile (f32 -> f16) into LDS, coalesced 256B per 4 threads
      const float* kr = K + (size_t)(j0 + jc) * D_DIM + dg * 16;
      if (j0 + 32 < S_DIM) __builtin_prefetch(kr + 32 * D_DIM, 0, 1);
      float4 x0 = *(const float4*)(kr + 0), x1 = *(const float4*)(kr + 4);
      float4 x2 = *(const float4*)(kr + 8), x3 = *(const float4*)(kr + 12);
      *(v8h*)(sK + jc * KSTR + dg * 16 + 0) = pack8(x0, x1);
      *(v8h*)(sK + jc * KSTR + dg * 16 + 8) = pack8(x2, x3);
    }
    __syncthreads();
#pragma unroll
    for (int s = 0; s < 2; ++s) {              // two 16-key score subtiles
      v8f sc = {};
#pragma unroll
      for (int c = 0; c < 2; ++c) {            // d chunks 0..31 / 32..63
        // B operand, same k order as A (32x16): lane<16 col=m K=0..15; lane>=16 col=m K=16..31
        const _Float16* kb = sK + (s * 16 + m) * KSTR + c * 32 + hi * 16;
        v16h B = cat16(*(const v8h*)(kb), *(const v8h*)(kb + 8));
        sc = wmma_f16(Qa[c], B, sc);
      }
#pragma unroll
      for (int e = 0; e < 8; ++e) {
        float x = fmaf(sc[e], SCALE, C_ADD);
        den[e] += x * x;
      }
    }
    __syncthreads();
  }

  // ---- reduce den across the 16 lanes of each half (C layout: row M=e+hi*8) ----
  float invden[8];
#pragma unroll
  for (int e = 0; e < 8; ++e) {
    float t = den[e];
#pragma unroll
    for (int off = 1; off < 16; off <<= 1) t += __shfl_xor(t, off, 32);
    invden[e] = 1.0f / (t + EPSV);
  }

  // ================= Sweep 2: prob out-stream + P@V =================
  v8f acc[4] = {};                             // out tile 16x64 (4 d-subtiles)
  for (int j0 = 0; j0 < S_DIM; j0 += 32) {
    {  // K tile
      const float* kr = K + (size_t)(j0 + jc) * D_DIM + dg * 16;
      if (j0 + 32 < S_DIM) __builtin_prefetch(kr + 32 * D_DIM, 0, 1);
      float4 x0 = *(const float4*)(kr + 0), x1 = *(const float4*)(kr + 4);
      float4 x2 = *(const float4*)(kr + 8), x3 = *(const float4*)(kr + 12);
      *(v8h*)(sK + jc * KSTR + dg * 16 + 0) = pack8(x0, x1);
      *(v8h*)(sK + jc * KSTR + dg * 16 + 8) = pack8(x2, x3);
    }
    {  // V tile, stored transposed [d][j] for j-contiguous B operands
      const float* vr = V + (size_t)(j0 + jc) * D_DIM + dg * 16;
      if (j0 + 32 < S_DIM) __builtin_prefetch(vr + 32 * D_DIM, 0, 1);
      float4 x0 = *(const float4*)(vr + 0), x1 = *(const float4*)(vr + 4);
      float4 x2 = *(const float4*)(vr + 8), x3 = *(const float4*)(vr + 12);
      float f[16] = { x0.x, x0.y, x0.z, x0.w, x1.x, x1.y, x1.z, x1.w,
                      x2.x, x2.y, x2.z, x2.w, x3.x, x3.y, x3.z, x3.w };
#pragma unroll
      for (int kk = 0; kk < 16; ++kk)
        sV[(dg * 16 + kk) * VSTR + jc] = (_Float16)f[kk];
    }
    __syncthreads();

#pragma unroll
    for (int s = 0; s < 2; ++s) {
      v8f sc = {};
#pragma unroll
      for (int c = 0; c < 2; ++c) {
        const _Float16* kb = sK + (s * 16 + m) * KSTR + c * 32 + hi * 16;
        v16h B = cat16(*(const v8h*)(kb), *(const v8h*)(kb + 8));
        sc = wmma_f16(Qa[c], B, sc);
      }
#pragma unroll
      for (int e = 0; e < 8; ++e) {
        float x = fmaf(sc[e], SCALE, C_ADD);
        float p = (x * x) * invden[e];
        const int M = e + hi * 8;                        // C layout row
        PR[(size_t)(q0 + M) * S_DIM + j0 + s * 16 + m] = p;   // stream prob (f32)
        sP[(w * 16 + M) * VSTR + s * 16 + m] = (_Float16)p;   // stage for transpose
      }
    }

    // Wave-local LDS RAW fence: prob scratch written by this wave, read below.
    asm volatile("s_wait_dscnt 0" ::: "memory");

    // prob tile -> A layout (16x32 over the 32 keys of this tile)
    v16h Pa;
    {
      const _Float16* pr = sP + (w * 16 + m) * VSTR + hi * 16;
      Pa = cat16(*(const v8h*)(pr), *(const v8h*)(pr + 8));
    }
#pragma unroll
    for (int t = 0; t < 4; ++t) {              // out d-subtiles 0..3
      const _Float16* vb = sV + (t * 16 + m) * VSTR + hi * 16;
      v16h B = cat16(*(const v8h*)(vb), *(const v8h*)(vb + 8));
      acc[t] = wmma_f16(Pa, B, acc[t]);
    }
    __syncthreads();
  }

  // ---- store out tile ----
#pragma unroll
  for (int t = 0; t < 4; ++t)
#pragma unroll
    for (int e = 0; e < 8; ++e)
      OUT[(size_t)(q0 + e + hi * 8) * D_DIM + t * 16 + m] = acc[t][e];
}

extern "C" void kernel_launch(void* const* d_in, const int* in_sizes, int n_in,
                              void* d_out, int out_size, void* d_ws, size_t ws_size,
                              hipStream_t stream) {
  (void)in_sizes; (void)n_in; (void)d_ws; (void)ws_size; (void)out_size;
  const float* q = (const float*)d_in[0];
  const float* k = (const float*)d_in[1];
  const float* v = (const float*)d_in[2];
  float* out  = (float*)d_out;                                   // [B,H,S,D]
  float* prob = out + (size_t)2 * 16 * S_DIM * D_DIM;            // [B,H,S,S]
  dim3 grid(S_DIM / 64, 2 * 16);   // 32 query-blocks x 32 (b,h)
  QuadAttention_kernel<<<grid, dim3(128), 0, stream>>>(q, k, v, out, prob);
}
